// Trident_89799176225718
// MI455X (gfx1250) — matrix operand order repaired and checked by reading, not compiled, not run on hardware
//
#include <hip/hip_runtime.h>

typedef __attribute__((ext_vector_type(16))) __bf16 v16bf;
typedef __attribute__((ext_vector_type(8)))  __bf16 v8bf;
typedef __attribute__((ext_vector_type(8)))  float  v8f;

#define TEMP_F  0.05f
#define ALPHA_F 0.1f
#define EPS_F   1e-12f

// ---------------------------------------------------------------------------
// Kernel 1: row-wise L2 normalize, f32 -> bf16.  One wave per row.
// ---------------------------------------------------------------------------
__global__ void k_normalize(const float* __restrict__ in, __bf16* __restrict__ out,
                            int rows, int D) {
  const int wave = threadIdx.x >> 5;
  const int lane = threadIdx.x & 31;
  const int row  = blockIdx.x * (blockDim.x >> 5) + wave;
  if (row >= rows) return;
  const float* p = in + (size_t)row * D;
  float ss = 0.f;
  for (int k = lane; k < D; k += 32) { float v = p[k]; ss = fmaf(v, v, ss); }
  #pragma unroll
  for (int off = 16; off >= 1; off >>= 1) ss += __shfl_xor(ss, off, 32);
  const float scale = 1.0f / fmaxf(sqrtf(ss), EPS_F);
  __bf16* q = out + (size_t)row * D;
  for (int k = lane; k < D; k += 32) q[k] = (__bf16)(p[k] * scale);
}

// ---------------------------------------------------------------------------
// Kernel 2: WMMA GEMM (f_norm @ w_norm^T / TEMP) fused with per-row sum-exp
// and argmax.  Geometry hardcoded: D=512 (16 K-chunks of 32), N walked in
// 25 blocks of 5 tiles, each wave owns 2 M-tiles (32 rows).
// All fragment loads use ONE A base and ONE B base pointer with compile-time
// byte offsets (fits 24-bit IOFFSET), so the hot loop has no 64-bit address
// VALU and no spills.  Softmax/argmax state is per-lane (|pred| <= 20 so
// direct sum-of-exp is f32-safe), merged across lanes once at the end.
// ---------------------------------------------------------------------------
#define KD 512            // embedding dim (compile-time for offset folding)
#define NBLK 5            // N-tiles per block
#define MT 2              // M-tiles per wave

__global__ void __launch_bounds__(128, 1)
k_wmma_lse(const __bf16* __restrict__ fA, const __bf16* __restrict__ wB,
           float* __restrict__ lse, int* __restrict__ amax,
           int mPairs, int C) {
  const int wave = threadIdx.x >> 5;
  const int lane = threadIdx.x & 31;
  const int task = blockIdx.x * (blockDim.x >> 5) + wave;
  if (task >= mPairs) return;          // wave-uniform: EXEC all-ones for WMMA
  const int mBase = task * (16 * MT);
  const int col = lane & 15;
  const int hi  = lane >> 4;

  // A-fragment layout: lane (m%16, half hi) holds 16 contiguous K at
  // k + 16*hi of row m.  M-tile t is a constant +t*16*KD elements.
  const __bf16* Abase = fA + (size_t)(mBase + col) * KD + (hi * 16);

  float sumE[MT][8], bestV[MT][8];
  int   bestN[MT][8];
  #pragma unroll
  for (int t = 0; t < MT; ++t)
    #pragma unroll
    for (int r = 0; r < 8; ++r) { sumE[t][r] = 0.f; bestV[t][r] = -INFINITY; bestN[t][r] = 0; }

  const float invT = 1.0f / TEMP_F;
  const int nBlocks = C / (16 * NBLK);       // 2000 / 80 = 25, exact
  // B-fragment layout (same K order as A): lane (n%16, half hi) holds 16 contiguous K at
  // k + 16*hi of w-row n.  N-tile j is a constant +j*16*KD elements.
  const __bf16* Bbase = wB + (size_t)col * KD + (hi * 16);

  for (int nB = 0; nB < nBlocks; ++nB) {
    const int nBase = nB * (16 * NBLK);
    const __bf16* B = Bbase + (size_t)nBase * KD;
    if (nB + 1 < nBlocks) __builtin_prefetch(B + (size_t)(16 * NBLK) * KD, 0, 1);

    v8f acc[MT][NBLK];
    #pragma unroll
    for (int t = 0; t < MT; ++t)
      #pragma unroll
      for (int j = 0; j < NBLK; ++j) acc[t][j] = (v8f){};

    #pragma unroll 4
    for (int k = 0; k < KD; k += 32) {
      union { v16bf v; v8bf h[2]; } au[MT];
      #pragma unroll
      for (int t = 0; t < MT; ++t) {
        au[t].h[0] = *(const v8bf*)(Abase + t * (16 * KD) + k);
        au[t].h[1] = *(const v8bf*)(Abase + t * (16 * KD) + k + 8);
      }
      #pragma unroll
      for (int j = 0; j < NBLK; ++j) {
        v16bf b = *(const v16bf*)(B + j * (16 * KD) + k);
        #pragma unroll
        for (int t = 0; t < MT; ++t)
          acc[t][j] = __builtin_amdgcn_wmma_f32_16x16x32_bf16(
              false, au[t].v, false, b, (short)0, acc[t][j], false, false);
      }
    }

    // Per-lane online update.  C layout: VGPR r of tile (t,j) holds row
    // (mBase + 16*t + r + 8*hi), column (nBase + 16*j + col).
    #pragma unroll
    for (int j = 0; j < NBLK; ++j) {
      const int n = nBase + j * 16 + col;
      #pragma unroll
      for (int t = 0; t < MT; ++t)
        #pragma unroll
        for (int r = 0; r < 8; ++r) {
          const float v = acc[t][j][r] * invT;
          sumE[t][r] += __expf(v);           // |v|<=20 -> no overflow in f32
          if (v > bestV[t][r]) { bestV[t][r] = v; bestN[t][r] = n; }
        }
    }
  }

  // Single cross-lane merge (xor 1,2,4,8 stays within each 16-lane half).
  #pragma unroll
  for (int t = 0; t < MT; ++t)
    #pragma unroll
    for (int r = 0; r < 8; ++r) {
      float s = sumE[t][r];
      #pragma unroll
      for (int off = 1; off < 16; off <<= 1) s += __shfl_xor(s, off, 32);
      float bv = bestV[t][r];
      int   bn = bestN[t][r];
      #pragma unroll
      for (int off = 1; off < 16; off <<= 1) {
        const float ov = __shfl_xor(bv, off, 32);
        const int   on = __shfl_xor(bn, off, 32);
        if (ov > bv || (ov == bv && on < bn)) { bv = ov; bn = on; }
      }
      if (col == 0) {                 // lanes 0 and 16 write their 8 rows each
        const int m = mBase + t * 16 + r + hi * 8;
        lse[m]  = __logf(s);
        amax[m] = bn;
      }
    }
}

// ---------------------------------------------------------------------------
// Kernel 3: per-row target + aux dot products (6 dots of length D), combine
// with lse/argmax, atomically accumulate means into out[0]=loss, out[1]=acc.
// ---------------------------------------------------------------------------
__global__ void k_row_loss(const __bf16* __restrict__ fA, const __bf16* __restrict__ wB,
                           const int* __restrict__ label, const int* __restrict__ aux,
                           const float* __restrict__ lse, const int* __restrict__ amax,
                           float* __restrict__ out, int bs, int D, int n_aux) {
  const int wave = threadIdx.x >> 5;
  const int lane = threadIdx.x & 31;
  const int row  = blockIdx.x * (blockDim.x >> 5) + wave;
  if (row >= bs) return;
  const __bf16* f = fA + (size_t)row * D;
  const int lbl = label[row];
  const float invT = 1.0f / TEMP_F;
  float tgt = 0.f, auxsum = 0.f;
  for (int j = -1; j < n_aux; ++j) {
    const int c = (j < 0) ? lbl : aux[row * n_aux + j];
    const __bf16* w = wB + (size_t)c * D;
    float d = 0.f;
    for (int k = lane; k < D; k += 32) d = fmaf((float)f[k], (float)w[k], d);
    #pragma unroll
    for (int off = 16; off >= 1; off >>= 1) d += __shfl_xor(d, off, 32);
    d *= invT;
    if (j < 0) tgt = d; else auxsum += d;
  }
  if (lane == 0) {
    // 0.5*(loss_normal + loss_ls) = lse - (1 - a/2)*tgt - (a/(2*n_aux))*sum_aux
    const float li = lse[row] - (1.0f - 0.5f * ALPHA_F) * tgt
                              - (0.5f * ALPHA_F / (float)n_aux) * auxsum;
    const float ai = (amax[row] == lbl) ? 1.0f : 0.0f;
    const float inv = 1.0f / (float)bs;
    atomicAdd(out + 0, li * inv);
    atomicAdd(out + 1, ai * inv);
  }
}

__global__ void k_init_out(float* __restrict__ out) {
  if (threadIdx.x < 2) out[threadIdx.x] = 0.0f;
}

// ---------------------------------------------------------------------------
extern "C" void kernel_launch(void* const* d_in, const int* in_sizes, int n_in,
                              void* d_out, int out_size, void* d_ws, size_t ws_size,
                              hipStream_t stream) {
  const float* features = (const float*)d_in[0];
  const float* wemb     = (const float*)d_in[1];
  const int*   label    = (const int*)d_in[2];
  const int*   aux      = (const int*)d_in[3];
  float*       out      = (float*)d_out;

  const int bs    = in_sizes[2];            // 16384
  const int n_aux = in_sizes[3] / bs;       // 5
  const int D     = in_sizes[0] / bs;       // 512 (== KD, hardcoded in kernel 2)
  const int C     = in_sizes[1] / D;        // 2000

  // workspace layout
  __bf16* fA  = (__bf16*)d_ws;                       // bs*D bf16   (16 MB)
  __bf16* wB  = fA + (size_t)bs * D;                 // C*D bf16    (2 MB)
  float*  lse = (float*)(wB + (size_t)C * D);        // bs f32
  int*    amx = (int*)(lse + bs);                    // bs i32

  // 1) normalize to bf16 (8 waves / block, one wave per row)
  k_normalize<<<(bs + 7) / 8, 256, 0, stream>>>(features, fA, bs, D);
  k_normalize<<<(C + 7) / 8, 256, 0, stream>>>(wemb, wB, C, D);

  // 2) fused WMMA GEMM + sum-exp + argmax: 2 M-tiles x (25 x 5 N-tiles)/wave
  const int mPairs = bs / 32;               // 512 wave-tasks
  k_wmma_lse<<<(mPairs + 3) / 4, 128, 0, stream>>>(fA, wB, lse, amx, mPairs, C);

  // 3) gather dots + loss/acc reduction
  k_init_out<<<1, 32, 0, stream>>>(out);
  k_row_loss<<<(bs + 7) / 8, 256, 0, stream>>>(fA, wB, label, aux, lse, amx,
                                               out, bs, D, n_aux);
}
